// GAGRUCell_36644660969996
// MI455X (gfx1250) — matrix operand order repaired and checked by reading, not compiled, not run on hardware
//
#include <hip/hip_runtime.h>
#include <hip/hip_bf16.h>

#define NN   4096
#define UU   64
#define INPD 2
#define DIMX 66
#define HH   4
#define HS   80          // h row stride padded to 5 WMMA tiles of 16
#define NW   (NN / 32)   // packed-adj words per row (128)
#define GR   8           // rows per block in the small GEMM
#define ATB  256         // attention block threads (8 waves)
#define ROWSPB (ATB / 32 * 16)   // 128 rows per attention block
#define JSPLIT 4         // j-range split for the single-layer passes
#define NEG_BIG   (-9.0e15f)
#define ALPHA_LR  0.2f
#define SLOPE_LR  0.01f

typedef __attribute__((ext_vector_type(16))) _Float16 v16h;
typedef __attribute__((ext_vector_type(8)))  float    v8f;

// ---------------------------------------------------------------------------
// Pack adjacency (0/1 int32) into bit words: padj[row*NW + w] bit b <-> adj[row][w*32+b]
// 67 MB -> 2 MB, read 10x by the attention passes (L2-resident).
// ---------------------------------------------------------------------------
__global__ void pack_adj_kernel(const int* __restrict__ adj,
                                unsigned* __restrict__ padj)
{
  int idx = blockIdx.x * blockDim.x + threadIdx.x;   // over NN*NW words
  if (idx >= NN * NW) return;
  size_t base = (size_t)idx * 32;
  unsigned w = 0;
  #pragma unroll
  for (int b = 0; b < 32; ++b)
    w |= (adj[base + b] > 0 ? 1u : 0u) << b;
  padj[idx] = w;
}

// ---------------------------------------------------------------------------
// Fused GAT attention: out = leakyrelu_{0.01}( softmax_row(masked e) @ h )
// e_ij = leakyrelu_{0.2}(f1_i + f2_j), mask from packed adj bits -> -9e15.
// Online (flash) softmax; attn@h via v_wmma_f32_16x16x32_f16.
// Block = ATB threads = 8 waves; wave w owns rows [blk*ROWSPB + w*16, +16).
// Double-buffered LDS staging of the 32x80 Hj tile (fp32 -> f16, transposed).
// blockIdx.z selects a j-chunk (jsplit>1 -> emit raw partials to pacc/pm/pl).
// ---------------------------------------------------------------------------
__global__ __launch_bounds__(ATB) void gat_attn_kernel(
    const float* __restrict__ hbase, size_t hHeadStride,
    const float* __restrict__ f1base, const float* __restrict__ f2base,
    size_t fHeadStride,
    const unsigned* __restrict__ padj,
    float* __restrict__ outp, int ldo, int colBase, int colHeadStep,
    int jsplit,
    float* __restrict__ pacc, float* __restrict__ pm, float* __restrict__ pl)
{
  const int head = blockIdx.y;
  const int z    = blockIdx.z;
  const float* __restrict__ hm  = hbase  + (size_t)head * hHeadStride;
  const float* __restrict__ f1v = f1base + (size_t)head * fHeadStride;
  const float* __restrict__ f2v = f2base + (size_t)head * fHeadStride;
  const int colOff = colBase + head * colHeadStep;
  const int jBeg = z * (NN / jsplit);
  const int jEnd = jBeg + (NN / jsplit);

  __shared__ float f2s[NN];                             // 16 KB
  __shared__ __align__(64) _Float16 HjT[2][HS * 32];    // 2 x 5 KB, [feature][k]

  const int tid  = threadIdx.x;
  const int wave = tid >> 5;
  const int lane = tid & 31;
  const int n    = lane & 15;   // col within 16-tile / row within M-tile
  const int hi   = lane >> 4;   // K-half selector

  for (int k = jBeg + tid; k < jEnd; k += ATB) f2s[k] = f2v[k];

  const int    rowBase = blockIdx.x * ROWSPB + wave * 16;
  const int    row     = rowBase + n;
  const float  f1r     = f1v[row];
  const size_t prowB   = (size_t)row * NW;

  float m_i = -3.0e38f;
  float l_i = 0.0f;
  v8f acc[5] = {};

  // stage Hj tile for j0 into LDS buffer `buf` (float4 reads, f16 transposed)
  auto stage = [&](int buf, int j0) {
    for (int idx = tid; idx < 32 * (HS / 4); idx += ATB) {
      int kr = idx / (HS / 4);
      int f4 = idx - kr * (HS / 4);
      const float4 v = *(const float4*)&hm[(size_t)(j0 + kr) * HS + f4 * 4];
      int ft = f4 * 4;
      HjT[buf][(ft + 0) * 32 + kr] = (_Float16)v.x;
      HjT[buf][(ft + 1) * 32 + kr] = (_Float16)v.y;
      HjT[buf][(ft + 2) * 32 + kr] = (_Float16)v.z;
      HjT[buf][(ft + 3) * 32 + kr] = (_Float16)v.w;
    }
  };

  stage(0, jBeg);
  int cur = 0;
  __syncthreads();

  for (int j0 = jBeg; j0 < jEnd; j0 += 32) {
    if (j0 + 32 < jEnd) stage(cur ^ 1, j0 + 32);   // prefetch next tile

    // one packed-adj word covers this lane's whole 32-col tile for its row
    const unsigned wbits = padj[prowB + (j0 >> 5)];

    // e values in contiguous A-operand k order (B tile uses the same):
    // lane<16: K in {0..15}; lane>=16: K in {16..31}
    float ev[16];
    float tmax = -3.0e38f;
    #pragma unroll
    for (int e = 0; e < 16; ++e) {
      int k = e + hi * 16;
      float x = f1r + f2s[j0 + k];
      x = x > 0.0f ? x : ALPHA_LR * x;
      x = ((wbits >> k) & 1u) ? x : NEG_BIG;
      ev[e] = x;
      tmax  = fmaxf(tmax, x);
    }
    tmax = fmaxf(tmax, __shfl_xor(tmax, 16));

    float mnew  = fmaxf(m_i, tmax);
    float alpha = __expf(m_i - mnew);
    m_i = mnew;

    float rsum = 0.0f;
    v16h aop;
    #pragma unroll
    for (int e = 0; e < 16; ++e) {
      float p = __expf(ev[e] - mnew);      // masked entries underflow to 0
      rsum += p;
      aop[e] = (_Float16)p;
    }
    rsum += __shfl_xor(rsum, 16);
    l_i = l_i * alpha + rsum;

    // rescale accumulators: C element r belongs to row r + 8*hi
    float al[8];
    #pragma unroll
    for (int r = 0; r < 8; ++r) al[r] = __shfl(alpha, r + hi * 8);
    #pragma unroll
    for (int t = 0; t < 5; ++t)
      #pragma unroll
      for (int r = 0; r < 8; ++r) acc[t][r] *= al[r];

    // 5 WMMA tiles: P(16x32 f16) @ Hj(32x16 f16) += C(16x16 f32)
    #pragma unroll
    for (int t = 0; t < 5; ++t) {
      v16h bop = *(const v16h*)&HjT[cur][(t * 16 + n) * 32 + hi * 16];
      acc[t] = __builtin_amdgcn_wmma_f32_16x16x32_f16(
          false, aop, false, bop, (short)0, acc[t], false, false);
    }

    __syncthreads();   // staged buffer visible; old buffer free for restage
    cur ^= 1;
  }

  if (jsplit == 1) {
    // finalize: divide by row sums, fused leaky-relu(0.01), write cols < 66
    float linv[8];
    #pragma unroll
    for (int r = 0; r < 8; ++r) linv[r] = 1.0f / __shfl(l_i, r + hi * 8);
    #pragma unroll
    for (int t = 0; t < 5; ++t) {
      int col = t * 16 + n;
      if (col < DIMX) {
        #pragma unroll
        for (int r = 0; r < 8; ++r) {
          int   orow = rowBase + r + hi * 8;
          float v = acc[t][r] * linv[r];
          v = v > 0.0f ? v : SLOPE_LR * v;
          outp[(size_t)orow * ldo + colOff + col] = v;
        }
      }
    }
  } else {
    // emit raw partials for this j-chunk; combine kernel merges them
    if (hi == 0) {
      pm[(size_t)z * NN + row] = m_i;
      pl[(size_t)z * NN + row] = l_i;
    }
    #pragma unroll
    for (int t = 0; t < 5; ++t) {
      int col = t * 16 + n;
      #pragma unroll
      for (int r = 0; r < 8; ++r) {
        int orow = rowBase + r + hi * 8;
        pacc[((size_t)z * NN + orow) * HS + col] = acc[t][r];
      }
    }
  }
}

// ---------------------------------------------------------------------------
// Merge JSPLIT flash-attention partials: out = leaky( (sum_s acc_s*w_s) / (sum_s l_s*w_s) ),
// w_s = exp(m_s - max_s m_s). All-masked chunks get w_s = 0 and vanish.
// ---------------------------------------------------------------------------
__global__ void attn_combine_kernel(const float* __restrict__ pacc,
                                    const float* __restrict__ pm,
                                    const float* __restrict__ pl,
                                    float* __restrict__ outp, int ldo)
{
  int idx = blockIdx.x * blockDim.x + threadIdx.x;
  if (idx >= NN * DIMX) return;
  int i = idx / DIMX, c = idx - i * DIMX;
  float mstar = -3.0e38f;
  #pragma unroll
  for (int s = 0; s < JSPLIT; ++s) mstar = fmaxf(mstar, pm[(size_t)s * NN + i]);
  float lsum = 0.0f, asum = 0.0f;
  #pragma unroll
  for (int s = 0; s < JSPLIT; ++s) {
    float w = __expf(pm[(size_t)s * NN + i] - mstar);
    lsum = fmaf(pl[(size_t)s * NN + i], w, lsum);
    asum = fmaf(pacc[((size_t)s * NN + i) * HS + c], w, asum);
  }
  float v = asum / lsum;
  v = v > 0.0f ? v : SLOPE_LR * v;
  outp[(size_t)i * ldo + c] = v;
}

// ---------------------------------------------------------------------------
// Small GEMM, GR rows per block: Out[i] = act(A[i] @ B + bias);
// optionally f1/f2 = h . a1/a2. act: 0=none,1=leaky(0.01),2=sigmoid,3=tanh.
// B is read once per 8 rows (8 accumulators per thread) to cut L2 traffic.
// ---------------------------------------------------------------------------
__global__ __launch_bounds__(128) void gemm_rows_kernel(
    const float* __restrict__ A, int lda,
    const float* __restrict__ B, size_t bHeadStride, int ldb,
    const float* __restrict__ bias,
    float* __restrict__ Out, size_t oHeadStride, int ldo,
    const float* __restrict__ a1, const float* __restrict__ a2, size_t avHeadStride,
    float* __restrict__ f1, float* __restrict__ f2, size_t fHeadStride,
    int K, int M, int act)
{
  __shared__ float arow[GR][264];
  __shared__ float hrow[GR][128];
  const int i0   = blockIdx.x * GR;
  const int head = blockIdx.y;
  const float* __restrict__ Bh = B + (size_t)head * bHeadStride;

  for (int idx = threadIdx.x; idx < GR * K; idx += 128) {
    int r = idx / K, k = idx - r * K;
    arow[r][k] = A[(size_t)(i0 + r) * lda + k];
  }
  __syncthreads();

  const int c = threadIdx.x;
  if (c < M) {
    float accr[GR];
    #pragma unroll
    for (int r = 0; r < GR; ++r) accr[r] = 0.0f;
    for (int k = 0; k < K; ++k) {
      float bk = Bh[(size_t)k * ldb + c];
      #pragma unroll
      for (int r = 0; r < GR; ++r) accr[r] = fmaf(arow[r][k], bk, accr[r]);
    }
    float bb = bias ? bias[c] : 0.0f;
    #pragma unroll
    for (int r = 0; r < GR; ++r) {
      float v = accr[r] + bb;
      if      (act == 1) v = v > 0.0f ? v : SLOPE_LR * v;
      else if (act == 2) v = 1.0f / (1.0f + __expf(-v));
      else if (act == 3) v = tanhf(v);
      Out[(size_t)head * oHeadStride + (size_t)(i0 + r) * ldo + c] = v;
      hrow[r][c] = v;
    }
  }
  __syncthreads();

  if (f1 != nullptr && threadIdx.x < GR) {
    const int r = threadIdx.x;
    const float* a1h = a1 + (size_t)head * avHeadStride;
    const float* a2h = a2 + (size_t)head * avHeadStride;
    float s1 = 0.0f, s2 = 0.0f;
    for (int c2 = 0; c2 < M; ++c2) {
      s1 = fmaf(hrow[r][c2], a1h[c2], s1);
      s2 = fmaf(hrow[r][c2], a2h[c2], s2);
    }
    f1[(size_t)head * fHeadStride + i0 + r] = s1;
    f2[(size_t)head * fHeadStride + i0 + r] = s2;
  }
}

// ---------------------------------------------------------------------------
// Elementwise glue kernels
// ---------------------------------------------------------------------------
__global__ void build_x1_kernel(const float* __restrict__ inputs,
                                const float* __restrict__ hx,
                                float* __restrict__ x1)
{
  int idx = blockIdx.x * blockDim.x + threadIdx.x;
  if (idx >= NN * DIMX) return;
  int i = idx / DIMX, c = idx - i * DIMX;
  x1[idx] = (c < INPD) ? inputs[i * INPD + c] : hx[i * UU + (c - INPD)];
}

__global__ void build_x2_kernel(const float* __restrict__ inputs,
                                const float* __restrict__ hx,
                                const float* __restrict__ value,
                                float* __restrict__ x2)
{
  int idx = blockIdx.x * blockDim.x + threadIdx.x;
  if (idx >= NN * DIMX) return;
  int i = idx / DIMX, c = idx - i * DIMX;
  x2[idx] = (c < INPD) ? inputs[i * INPD + c]
                       : value[i * 128 + (c - INPD)] * hx[i * UU + (c - INPD)];
}

__global__ void final_kernel(const float* __restrict__ hx,
                             const float* __restrict__ value,
                             const float* __restrict__ cbuf,
                             float* __restrict__ out)
{
  int idx = blockIdx.x * blockDim.x + threadIdx.x;
  if (idx >= NN * UU) return;
  int i = idx / UU, j = idx - i * UU;
  float u = value[i * 128 + UU + j];
  out[idx] = u * hx[idx] + (1.0f - u) * cbuf[idx];
}

// ---------------------------------------------------------------------------
extern "C" void kernel_launch(void* const* d_in, const int* in_sizes, int n_in,
                              void* d_out, int out_size, void* d_ws, size_t ws_size,
                              hipStream_t stream)
{
  const float* inputs  = (const float*)d_in[0];
  const float* hx      = (const float*)d_in[1];
  const int*   adj     = (const int*)  d_in[2];
  const float* m1_W    = (const float*)d_in[3];
  const float* m1_a1   = (const float*)d_in[4];
  const float* m1_a2   = (const float*)d_in[5];
  const float* m1_Wout = (const float*)d_in[6];
  const float* m1_ao1  = (const float*)d_in[7];
  const float* m1_ao2  = (const float*)d_in[8];
  const float* m2_W    = (const float*)d_in[9];
  const float* m2_a1   = (const float*)d_in[10];
  const float* m2_a2   = (const float*)d_in[11];
  const float* m2_Wout = (const float*)d_in[12];
  const float* m2_ao1  = (const float*)d_in[13];
  const float* m2_ao2  = (const float*)d_in[14];
  const float* g1_W    = (const float*)d_in[15];
  const float* g1_b    = (const float*)d_in[16];
  const float* g2_W    = (const float*)d_in[17];
  const float* g2_b    = (const float*)d_in[18];

  char* wsb = (char*)d_ws;
  size_t off = 0;
  auto alloc = [&](size_t bytes) -> void* {
    void* p = (void*)(wsb + off);
    off += (bytes + 255) & ~(size_t)255;
    return p;
  };

  unsigned* padj = (unsigned*)alloc((size_t)NN * NW * 4);
  float* x1    = (float*)alloc((size_t)NN * DIMX * 4);
  float* h1    = (float*)alloc((size_t)HH * NN * HS * 4);
  float* f11   = (float*)alloc((size_t)HH * NN * 4);
  float* f21   = (float*)alloc((size_t)HH * NN * 4);
  float* hcat  = (float*)alloc((size_t)NN * HH * DIMX * 4);
  float* hout1 = (float*)alloc((size_t)NN * HS * 4);
  float* f1o   = (float*)alloc((size_t)NN * 4);
  float* f2o   = (float*)alloc((size_t)NN * 4);
  float* g1buf = (float*)alloc((size_t)NN * HS * 4);
  float* value = (float*)alloc((size_t)NN * 128 * 4);
  float* x2    = (float*)alloc((size_t)NN * DIMX * 4);
  float* h2    = (float*)alloc((size_t)HH * NN * HS * 4);
  float* f12   = (float*)alloc((size_t)HH * NN * 4);
  float* f22   = (float*)alloc((size_t)HH * NN * 4);
  float* hcat2 = (float*)alloc((size_t)NN * HH * DIMX * 4);
  float* hout2 = (float*)alloc((size_t)NN * HS * 4);
  float* f1o2  = (float*)alloc((size_t)NN * 4);
  float* f2o2  = (float*)alloc((size_t)NN * 4);
  float* g2buf = (float*)alloc((size_t)NN * HS * 4);
  float* cbuf  = (float*)alloc((size_t)NN * UU * 4);
  float* pacc  = (float*)alloc((size_t)JSPLIT * NN * HS * 4);  // split-j partials
  float* pm    = (float*)alloc((size_t)JSPLIT * NN * 4);
  float* pl    = (float*)alloc((size_t)JSPLIT * NN * 4);

  const dim3 bAtt(ATB);
  const dim3 b128(128);
  const int ewBlocks = (NN * DIMX + 255) / 256;

  pack_adj_kernel<<<(NN * NW + 255) / 256, 256, 0, stream>>>(adj, padj);

  // ---- pass 1 -------------------------------------------------------------
  build_x1_kernel<<<ewBlocks, 256, 0, stream>>>(inputs, hx, x1);

  gemm_rows_kernel<<<dim3(NN / GR, HH), b128, 0, stream>>>(
      x1, DIMX, m1_W, (size_t)DIMX * DIMX, DIMX, nullptr,
      h1, (size_t)NN * HS, HS, m1_a1, m1_a2, DIMX,
      f11, f21, NN, DIMX, DIMX, 0);

  gat_attn_kernel<<<dim3(NN / ROWSPB, HH, 1), bAtt, 0, stream>>>(
      h1, (size_t)NN * HS, f11, f21, NN, padj,
      hcat, HH * DIMX, 0, DIMX, 1, nullptr, nullptr, nullptr);

  gemm_rows_kernel<<<dim3(NN / GR, 1), b128, 0, stream>>>(
      hcat, HH * DIMX, m1_Wout, 0, DIMX, nullptr,
      hout1, 0, HS, m1_ao1, m1_ao2, 0,
      f1o, f2o, 0, HH * DIMX, DIMX, 0);

  gat_attn_kernel<<<dim3(NN / ROWSPB, 1, JSPLIT), bAtt, 0, stream>>>(
      hout1, 0, f1o, f2o, 0, padj,
      nullptr, 0, 0, 0, JSPLIT, pacc, pm, pl);
  attn_combine_kernel<<<ewBlocks, 256, 0, stream>>>(pacc, pm, pl, g1buf, HS);

  gemm_rows_kernel<<<dim3(NN / GR, 1), b128, 0, stream>>>(
      g1buf, HS, g1_W, 0, 2 * UU, g1_b,
      value, 0, 2 * UU, nullptr, nullptr, 0,
      nullptr, nullptr, 0, DIMX, 2 * UU, 2);

  // ---- pass 2 -------------------------------------------------------------
  build_x2_kernel<<<ewBlocks, 256, 0, stream>>>(inputs, hx, value, x2);

  gemm_rows_kernel<<<dim3(NN / GR, HH), b128, 0, stream>>>(
      x2, DIMX, m2_W, (size_t)DIMX * DIMX, DIMX, nullptr,
      h2, (size_t)NN * HS, HS, m2_a1, m2_a2, DIMX,
      f12, f22, NN, DIMX, DIMX, 0);

  gat_attn_kernel<<<dim3(NN / ROWSPB, HH, 1), bAtt, 0, stream>>>(
      h2, (size_t)NN * HS, f12, f22, NN, padj,
      hcat2, HH * DIMX, 0, DIMX, 1, nullptr, nullptr, nullptr);

  gemm_rows_kernel<<<dim3(NN / GR, 1), b128, 0, stream>>>(
      hcat2, HH * DIMX, m2_Wout, 0, DIMX, nullptr,
      hout2, 0, HS, m2_ao1, m2_ao2, 0,
      f1o2, f2o2, 0, HH * DIMX, DIMX, 0);

  gat_attn_kernel<<<dim3(NN / ROWSPB, 1, JSPLIT), bAtt, 0, stream>>>(
      hout2, 0, f1o2, f2o2, 0, padj,
      nullptr, 0, 0, 0, JSPLIT, pacc, pm, pl);
  attn_combine_kernel<<<ewBlocks, 256, 0, stream>>>(pacc, pm, pl, g2buf, HS);

  gemm_rows_kernel<<<dim3(NN / GR, 1), b128, 0, stream>>>(
      g2buf, HS, g2_W, 0, UU, g2_b,
      cbuf, 0, UU, nullptr, nullptr, 0,
      nullptr, nullptr, 0, DIMX, UU, 3);

  final_kernel<<<(NN * UU + 255) / 256, 256, 0, stream>>>(
      hx, value, cbuf, (float*)d_out);

  (void)in_sizes; (void)n_in; (void)out_size; (void)ws_size;
}
